// DecoderGRU_28819230556832
// MI455X (gfx1250) — matrix operand order repaired and checked by reading, compiled, not run on hardware
//
#include <hip/hip_runtime.h>

typedef __attribute__((ext_vector_type(16))) _Float16 v16h;
typedef __attribute__((ext_vector_type(8)))  _Float16 v8h;
typedef __attribute__((ext_vector_type(8)))  float    v8f;

#define BSZ  32768
#define HID  128
#define INP  64
#define SPANC 8
#define OUTC 2
#define DAYS 20

// LDS byte offsets (all 32B aligned)
#define OFF_WIH   0          // 384 x 64  f16
#define OFF_WHH   49152      // 384 x 128 f16
#define OFF_FIN   147456     // 64 x 128  f16
#define OFF_FOUT  163840     // 16 x 128  f16 (col 0 = w0-w1, rest zero)
#define OFF_SPAN  167936     // 16 x 128  f16 (rows 8..15 zero)
#define OFF_HBUF  172032     // 128 x 128 f16
#define OFF_XBUF  204800     // 128 x 64  f16
#define OFF_BRZ   221184     // 256 f32 (b_ih+b_hh for r,z)
#define OFF_BIN   222208     // 128 f32
#define OFF_BHN   222720     // 128 f32
#define OFF_BX    223232     // 64  f32
#define OFF_BOUT  223488     // 16  f32 ([0] = b0-b1)
#define OFF_BSPAN 223552     // 16  f32
#define SMEM_BYTES 223616

__device__ __forceinline__ v8f wmma16(v16h a, v16h b, v8f c) {
    return __builtin_amdgcn_wmma_f32_16x16x32_f16(false, a, false, b, (short)0, c, false, false);
}
// B fragment: column-pointer (K-contiguous); elements 0..7 -> K=kb+e+8*hi, 8..15 -> +16
__device__ __forceinline__ v16h ldsB(const _Float16* col, int kb, int hi) {
    union { v16h v; v8h h[2]; } u;
    u.h[0] = *(const v8h*)(col + kb + hi * 8);
    u.h[1] = *(const v8h*)(col + kb + 16 + hi * 8);
    return u.v;
}
// A fragment: row-pointer [M][K]; elements 0..7 -> K=kb+e+8*hi, 8..15 -> +16
__device__ __forceinline__ v16h ldsA(const _Float16* row, int kb, int hi) {
    union { v16h v; v8h h[2]; } u;
    u.h[0] = *(const v8h*)(row + kb + hi * 8);
    u.h[1] = *(const v8h*)(row + kb + 16 + hi * 8);
    return u.v;
}
__device__ __forceinline__ v8f splat8(float x) {
    v8f v;
#pragma unroll
    for (int i = 0; i < 8; ++i) v[i] = x;
    return v;
}
__device__ __forceinline__ float sigm(float x) { return 1.0f / (1.0f + __expf(-x)); }
__device__ __forceinline__ float tanh_f(float x) { return 1.0f - 2.0f / (__expf(2.0f * x) + 1.0f); }

// ds_swizzle bit-mask mode: and=0x1F, or=0, xor=XM  (pattern = (XM<<10)|0x1F)
template <int PAT>
__device__ __forceinline__ float swz(float x) {
    return __int_as_float(__builtin_amdgcn_ds_swizzle(__float_as_int(x), PAT));
}

extern "C" __global__ __launch_bounds__(256)
void decoder_gru_kernel(const float* __restrict__ hx,
                        const float* __restrict__ w_ih,  const float* __restrict__ w_hh,
                        const float* __restrict__ b_ih,  const float* __restrict__ b_hh,
                        const float* __restrict__ fc_in_w, const float* __restrict__ fc_in_b,
                        const float* __restrict__ fc_out_w, const float* __restrict__ fc_out_b,
                        const float* __restrict__ span_w, const float* __restrict__ span_b,
                        float* __restrict__ out)
{
    extern __shared__ __align__(32) char smem[];
    _Float16* WihT  = (_Float16*)(smem + OFF_WIH);
    _Float16* WhhT  = (_Float16*)(smem + OFF_WHH);
    _Float16* FinT  = (_Float16*)(smem + OFF_FIN);
    _Float16* FoutT = (_Float16*)(smem + OFF_FOUT);
    _Float16* SpanT = (_Float16*)(smem + OFF_SPAN);
    _Float16* hbuf  = (_Float16*)(smem + OFF_HBUF);
    _Float16* xbuf  = (_Float16*)(smem + OFF_XBUF);
    float* brz   = (float*)(smem + OFF_BRZ);
    float* bin   = (float*)(smem + OFF_BIN);
    float* bhn   = (float*)(smem + OFF_BHN);
    float* bx    = (float*)(smem + OFF_BX);
    float* bout  = (float*)(smem + OFF_BOUT);
    float* bspan = (float*)(smem + OFF_BSPAN);

    const int tid = threadIdx.x;
    const int wgRow0 = blockIdx.x * 128;

    // ---- cooperative init: f32 -> f16 weights into LDS (already K-contig layout) ----
    for (int i = tid; i < 384 * 64;  i += 256) WihT[i] = (_Float16)w_ih[i];
    for (int i = tid; i < 384 * 128; i += 256) WhhT[i] = (_Float16)w_hh[i];
    for (int i = tid; i < 64 * 128;  i += 256) FinT[i] = (_Float16)fc_in_w[i];
    for (int i = tid; i < 16 * 128;  i += 256) {
        // col 0 (rows 0..127 of the K-major layout) = fc_out_w[0,:] - fc_out_w[1,:]
        _Float16 v = (_Float16)0.0f;
        if (i < 128) v = (_Float16)(fc_out_w[i] - fc_out_w[128 + i]);
        FoutT[i] = v;
    }
    for (int i = tid; i < 16 * 128;  i += 256) SpanT[i] = (i < SPANC * 128) ? (_Float16)span_w[i] : (_Float16)0.0f;
    for (int i = tid; i < 256; i += 256) brz[i] = b_ih[i] + b_hh[i];
    for (int i = tid; i < 128; i += 256) { bin[i] = b_ih[256 + i]; bhn[i] = b_hh[256 + i]; }
    for (int i = tid; i < 64;  i += 256) bx[i] = fc_in_b[i];
    for (int i = tid; i < 16;  i += 256) {
        bout[i]  = (i == 0)    ? (fc_out_b[0] - fc_out_b[1]) : 0.0f;
        bspan[i] = (i < SPANC) ? span_b[i] : 0.0f;
    }
    for (int i = tid; i < 128 * HID; i += 256) hbuf[i] = (_Float16)hx[(size_t)wgRow0 * HID + i];
    for (int i = tid; i < 128 * INP; i += 256) xbuf[i] = (_Float16)0.0f;
    __syncthreads();

    const int lane = tid & 31;
    const int wv   = tid >> 5;
    const int hi   = lane >> 4;      // lane group (0: lanes 0-15, 1: lanes 16-31)
    const int lo   = lane & 15;      // N column within tile / M row for A-frag

    const _Float16* hrow = hbuf + (wv * 16 + lo) * HID;   // A-frag row (M = lo)
    const _Float16* xrow = xbuf + (wv * 16 + lo) * INP;
    _Float16* hsl = hbuf + wv * 16 * HID;                 // wave's h slice
    _Float16* xsl = xbuf + wv * 16 * INP;

    v16h hA[4], xA[2];
#pragma unroll
    for (int c = 0; c < 4; ++c) hA[c] = ldsA(hrow, c * 32, hi);
#pragma unroll
    for (int c = 0; c < 2; ++c) xA[c] = ldsA(xrow, c * 32, hi);

    float* outSpan = out;
    float* outDays = out + (size_t)BSZ * SPANC;
    const int growBase = wgRow0 + wv * 16;

    // ---- num_spans = softmax(hx @ span_w.T + b), once ----
    {
        v8f s = splat8(bspan[lo]);
        const _Float16* col = SpanT + lo * HID;
#pragma unroll
        for (int c = 0; c < 4; ++c) s = wmma16(hA[c], ldsB(col, c * 32, hi), s);
#pragma unroll
        for (int r = 0; r < 8; ++r) {
            float v = s[r];
            float m = v;
            m = fmaxf(m, swz<0x041F>(m));   // xor 1
            m = fmaxf(m, swz<0x081F>(m));   // xor 2
            m = fmaxf(m, swz<0x101F>(m));   // xor 4
            float e = __expf(v - m);
            float t = e;
            t += swz<0x041F>(t);
            t += swz<0x081F>(t);
            t += swz<0x101F>(t);
            if (lo < SPANC)
                outSpan[(size_t)(growBase + r + hi * 8) * SPANC + lo] = e / t;
        }
    }

    const float boutDiff = bout[0];

    // ---- recurrent day loop ----
    for (int day = 0; day < DAYS; ++day) {
        // GRU gates, 8 column-tiles of HID
        for (int jt = 0; jt < 8; ++jt) {
            const int n = jt * 16 + lo;
            const _Float16* cihR = WihT + n * INP;
            const _Float16* cihZ = WihT + (128 + n) * INP;
            const _Float16* cihN = WihT + (256 + n) * INP;
            const _Float16* chhR = WhhT + n * HID;
            const _Float16* chhZ = WhhT + (128 + n) * HID;
            const _Float16* chhN = WhhT + (256 + n) * HID;

            v8f aR = splat8(brz[n]);
            v8f aZ = splat8(brz[128 + n]);
            v8f aI = splat8(bin[n]);
            v8f aH = splat8(bhn[n]);
#pragma unroll
            for (int c = 0; c < 2; ++c) {
                aR = wmma16(xA[c], ldsB(cihR, c * 32, hi), aR);
                aZ = wmma16(xA[c], ldsB(cihZ, c * 32, hi), aZ);
                aI = wmma16(xA[c], ldsB(cihN, c * 32, hi), aI);
            }
#pragma unroll
            for (int c = 0; c < 4; ++c) {
                aR = wmma16(hA[c], ldsB(chhR, c * 32, hi), aR);
                aZ = wmma16(hA[c], ldsB(chhZ, c * 32, hi), aZ);
                aH = wmma16(hA[c], ldsB(chhN, c * 32, hi), aH);
            }
#pragma unroll
            for (int r = 0; r < 8; ++r) {
                const int m = r + hi * 8;
                float hold = (float)hsl[m * HID + n];
                float rg = sigm(aR[r]);
                float zg = sigm(aZ[r]);
                float ng = tanh_f(aI[r] + rg * aH[r]);
                float hnew = (1.0f - zg) * ng + zg * hold;
                hsl[m * HID + n] = (_Float16)hnew;   // same location read above; wave-ordered DS
            }
        }
        // refresh h A-fragments for this day's heads + next day
#pragma unroll
        for (int c = 0; c < 4; ++c) hA[c] = ldsA(hrow, c * 32, hi);

        // fc_out: 2-class softmax == sigmoid of difference logit (col 0 of FoutT)
        {
            v8f o = splat8(boutDiff);
            const _Float16* col = FoutT + lo * HID;
#pragma unroll
            for (int c = 0; c < 4; ++c) o = wmma16(hA[c], ldsB(col, c * 32, hi), o);
#pragma unroll
            for (int r = 0; r < 8; ++r) {
                if (lo == 0) {
                    float p0 = sigm(o[r]);
                    float2 pv = make_float2(p0, 1.0f - p0);
                    *(float2*)(outDays + (size_t)(growBase + r + hi * 8) * (DAYS * OUTC) + day * OUTC) = pv;
                }
            }
        }

        // x = relu(h @ fc_in_w.T + b)
        for (int xt = 0; xt < 4; ++xt) {
            const int n = xt * 16 + lo;
            v8f a = splat8(bx[n]);
            const _Float16* col = FinT + n * HID;
#pragma unroll
            for (int c = 0; c < 4; ++c) a = wmma16(hA[c], ldsB(col, c * 32, hi), a);
#pragma unroll
            for (int r = 0; r < 8; ++r)
                xsl[(r + hi * 8) * INP + n] = (_Float16)fmaxf(a[r], 0.0f);
        }
#pragma unroll
        for (int c = 0; c < 2; ++c) xA[c] = ldsA(xrow, c * 32, hi);
    }
}

extern "C" void kernel_launch(void* const* d_in, const int* in_sizes, int n_in,
                              void* d_out, int out_size, void* d_ws, size_t ws_size,
                              hipStream_t stream) {
    (void)in_sizes; (void)n_in; (void)out_size; (void)d_ws; (void)ws_size;
    const float* hx       = (const float*)d_in[0];
    const float* w_ih     = (const float*)d_in[1];
    const float* w_hh     = (const float*)d_in[2];
    const float* b_ih     = (const float*)d_in[3];
    const float* b_hh     = (const float*)d_in[4];
    const float* fc_in_w  = (const float*)d_in[5];
    const float* fc_in_b  = (const float*)d_in[6];
    const float* fc_out_w = (const float*)d_in[7];
    const float* fc_out_b = (const float*)d_in[8];
    const float* span_w   = (const float*)d_in[9];
    const float* span_b   = (const float*)d_in[10];

    // Allow the large (224KB) dynamic LDS allocation; idempotent, non-stream op.
    static_assert(SMEM_BYTES <= 320 * 1024, "LDS budget");
    hipFuncSetAttribute((const void*)decoder_gru_kernel,
                        hipFuncAttributeMaxDynamicSharedMemorySize, SMEM_BYTES);

    dim3 grid(BSZ / 128), block(256);
    decoder_gru_kernel<<<grid, block, SMEM_BYTES, stream>>>(
        hx, w_ih, w_hh, b_ih, b_hh, fc_in_w, fc_in_b,
        fc_out_w, fc_out_b, span_w, span_b, (float*)d_out);
}
